// VectorQuantizer_4294967296503
// MI455X (gfx1250) — matrix operand order repaired and checked by reading, not compiled, not run on hardware
//
#include <hip/hip_runtime.h>

#define D_DIM   256
#define K_CODES 4096
#define N_PIX   65536            // 16*64*64
#define M_TILES (N_PIX / 16)     // 4096
#define K_TILES (K_CODES / 16)   // 256
#define TOTAL_Z (16 * 256 * 64 * 64) // 16777216

typedef __attribute__((ext_vector_type(16))) __bf16 v16bf;
typedef __attribute__((ext_vector_type(8)))  float  v8f;
typedef __attribute__((ext_vector_type(4)))  int    v4i;

#if __has_builtin(__builtin_amdgcn_global_load_async_to_lds_b128) && \
    __has_builtin(__builtin_amdgcn_s_wait_asynccnt)
#define USE_ASYNC_LDS 1
#else
#define USE_ASYNC_LDS 0
#endif

__device__ __forceinline__ unsigned short f32_to_bf16(float f) {
  unsigned int u = __float_as_uint(f);
  u += 0x7fffu + ((u >> 16) & 1u);       // round to nearest even
  return (unsigned short)(u >> 16);
}

// ---------------------------------------------------------------------------
// Pack z (B,D,H,W) -> bf16 A-operand tiles in contiguous-k A order (k = 16*(L>>4) + e).
// Tile t covers rows n = t*16 .. t*16+15 of z_flat (N x D).
// Per (lane L, step s): 16 bf16 at zA + t*4096 + s*512 + L*16.
//   L<16 : row m=L,   d = 32s + {0..15}
//   L>=16: row m=L-16,d = 32s + {16..31}
// ---------------------------------------------------------------------------
__global__ void vq_swizzle_a(const float* __restrict__ z,
                             unsigned short* __restrict__ zA) {
  int t  = blockIdx.x;
  int L  = threadIdx.x & 31;
  int s  = threadIdx.x >> 5;             // 0..7
  int m  = L & 15;
  int hi = L >> 4;
  int n  = t * 16 + m;
  int b  = n >> 12;                      // / 4096
  int hw = n & 4095;
  unsigned short* dst = zA + (size_t)t * 4096 + s * 512 + L * 16;
  const float* src = z + (size_t)b * (256 * 4096) + hw;
#pragma unroll
  for (int e = 0; e < 16; ++e) {
    int d = s * 32 + hi * 16 + e;
    dst[e] = f32_to_bf16(src[(size_t)d * 4096]);
  }
}

// ---------------------------------------------------------------------------
// Pack weight (K,D) -> bf16 B-operand tiles in matching contiguous-k B order.
// Per (lane L, step s): column = kt*16 + (L&15), d = 32s + 16*(L>>4) + e.
// ---------------------------------------------------------------------------
__global__ void vq_swizzle_b(const float* __restrict__ w,
                             unsigned short* __restrict__ wB) {
  int kt = blockIdx.x;
  int L  = threadIdx.x & 31;
  int s  = threadIdx.x >> 5;
  int r  = kt * 16 + (L & 15);
  int d0 = s * 32 + ((L >> 4) * 16);
  unsigned short* dst = wB + (size_t)kt * 4096 + s * 512 + L * 16;
  const float* src = w + (size_t)r * 256 + d0;
#pragma unroll
  for (int e = 0; e < 16; ++e) dst[e] = f32_to_bf16(src[e]);
}

// ---------------------------------------------------------------------------
// |w_k|^2 per code, one wave per row.
// ---------------------------------------------------------------------------
__global__ void vq_wnorm(const float* __restrict__ w, float* __restrict__ wnorm) {
  int wave = threadIdx.x >> 5, lane = threadIdx.x & 31;
  int k = blockIdx.x * 8 + wave;
  const float* row = w + (size_t)k * 256;
  float s = 0.f;
  for (int i = lane; i < 256; i += 32) { float v = row[i]; s += v * v; }
#pragma unroll
  for (int off = 16; off > 0; off >>= 1) s += __shfl_xor(s, off, 32);
  if (lane == 0) wnorm[k] = s;
}

// ---------------------------------------------------------------------------
// Stage one 8 KB B-tile into LDS: async DMA (ASYNCcnt path) when available,
// else load->store through VGPRs. Builtin signature (from the gfx1250
// diagnostic): (int4 addrspace(1)* src, lds dst, imm offset, imm cpol).
// ---------------------------------------------------------------------------
__device__ __forceinline__ void stage_b_tile(const unsigned short* __restrict__ src,
                                             unsigned short* dst, int tid) {
#if USE_ASYNC_LDS
  typedef __attribute__((address_space(1))) v4i GV;
  typedef __attribute__((address_space(3))) v4i LV;
  __builtin_amdgcn_global_load_async_to_lds_b128(
      (GV*)(src + (size_t)tid * 8), (LV*)(dst + (size_t)tid * 8), 0, 0);
  __builtin_amdgcn_global_load_async_to_lds_b128(
      (GV*)(src + (size_t)(tid + 256) * 8), (LV*)(dst + (size_t)(tid + 256) * 8), 0, 0);
#else
  ((uint4*)dst)[tid]       = ((const uint4*)src)[tid];
  ((uint4*)dst)[tid + 256] = ((const uint4*)src)[tid + 256];
#endif
}

__device__ __forceinline__ void stage_wait() {
#if USE_ASYNC_LDS
  __builtin_amdgcn_s_wait_asynccnt(0);
#endif
}

// ---------------------------------------------------------------------------
// Fused bf16-WMMA GEMM + argmin.
//  - Block = 8 waves; each wave owns TWO 16-row A strips in registers
//    (128 VGPRs), so each B fragment feeds 2 WMMAs.
//  - B k-tile (8 KB) staged cooperatively into a double-buffered LDS ring
//    shared by all 8 waves (async global->LDS when available).
//  - launch_bounds(256,2): ~512-VGPR cap -> no scratch spills.
// score(n,k) = |w_k|^2 - 2 * <z_n, w_k>   (|z_n|^2 row-constant -> dropped)
// ---------------------------------------------------------------------------
__global__ __launch_bounds__(256, 2) void vq_gemm_argmin(
    const unsigned short* __restrict__ zA,
    const unsigned short* __restrict__ wB,
    const float* __restrict__ wnorm,
    int* __restrict__ idx_out) {
  __shared__ unsigned short bsm[2][4096];        // 2 x 8 KB B-tile buffers

  int wave   = threadIdx.x >> 5;
  int lane   = threadIdx.x & 31;
  int tid    = threadIdx.x;
  int mtile0 = blockIdx.x * 16 + wave * 2;       // two m-tiles per wave

  // Load both 16x256 bf16 A-strips into registers (2 x 64 VGPRs).
  const unsigned short* aBase = zA + (size_t)mtile0 * 4096 + lane * 16;
  v16bf a0[8], a1[8];
#pragma unroll
  for (int s = 0; s < 8; ++s) a0[s] = *(const v16bf*)(aBase + s * 512);
#pragma unroll
  for (int s = 0; s < 8; ++s) a1[s] = *(const v16bf*)(aBase + 4096 + s * 512);

  float vmin[2][8];
  int   varg[2][8];
#pragma unroll
  for (int t = 0; t < 2; ++t)
#pragma unroll
    for (int j = 0; j < 8; ++j) { vmin[t][j] = 3.4e38f; varg[t][j] = 0; }

  stage_b_tile(wB, &bsm[0][0], tid);             // tile 0

  int col = lane & 15;                           // code within tile (C layout)

  for (int kt = 0; kt < K_TILES; ++kt) {
    int cur = kt & 1;
    stage_wait();                                // my async stores to LDS done
    __syncthreads();                             // all waves' stores visible
    if (kt + 1 < K_TILES)
      stage_b_tile(wB + (size_t)(kt + 1) * 4096, &bsm[cur ^ 1][0], tid);

    v8f c0 = {}, c1 = {};
    const unsigned short* bb = &bsm[cur][0] + lane * 16;
#pragma unroll
    for (int s = 0; s < 8; ++s) {
      v16bf b = *(const v16bf*)(bb + s * 512);
      c0 = __builtin_amdgcn_wmma_f32_16x16x32_bf16(
               false, a0[s], false, b, (short)0, c0, false, false);
      c1 = __builtin_amdgcn_wmma_f32_16x16x32_bf16(
               false, a1[s], false, b, (short)0, c1, false, false);
    }

    int k = kt * 16 + col;
    float wn = wnorm[k];
#pragma unroll
    for (int j = 0; j < 8; ++j) {
      float s0 = wn - 2.0f * c0[j];
      bool l0 = s0 < vmin[0][j];
      vmin[0][j] = l0 ? s0 : vmin[0][j];
      varg[0][j] = l0 ? k  : varg[0][j];
      float s1 = wn - 2.0f * c1[j];
      bool l1 = s1 < vmin[1][j];
      vmin[1][j] = l1 ? s1 : vmin[1][j];
      varg[1][j] = l1 ? k  : varg[1][j];
    }
  }

  // Reduce across the 16 lanes of each half-wave (rows 0..7 in lanes 0-15,
  // rows 8..15 in lanes 16-31). Tie-break: smallest index (matches argmin).
#pragma unroll
  for (int off = 1; off < 16; off <<= 1) {
#pragma unroll
    for (int t = 0; t < 2; ++t)
#pragma unroll
      for (int j = 0; j < 8; ++j) {
        float om = __shfl_xor(vmin[t][j], off, 32);
        int   oa = __shfl_xor(varg[t][j], off, 32);
        bool take = (om < vmin[t][j]) || (om == vmin[t][j] && oa < varg[t][j]);
        vmin[t][j] = take ? om : vmin[t][j];
        varg[t][j] = take ? oa : varg[t][j];
      }
  }
  if ((lane & 15) == 0) {
#pragma unroll
    for (int t = 0; t < 2; ++t) {
      int mbase = (mtile0 + t) * 16 + (lane >> 4) * 8;
#pragma unroll
      for (int j = 0; j < 8; ++j) idx_out[mbase + j] = varg[t][j];
    }
  }
}

// ---------------------------------------------------------------------------
// Zero the accumulators (dw, counts, scalars[2] = {n_total, loss_acc}).
// ---------------------------------------------------------------------------
__global__ void vq_init(float* __restrict__ dw, float* __restrict__ counts,
                        float* __restrict__ scalars) {
  int i = blockIdx.x * blockDim.x + threadIdx.x;
  int stride = gridDim.x * blockDim.x;
  for (int g = i; g < K_CODES * D_DIM; g += stride) dw[g] = 0.f;
  for (int g = i; g < K_CODES; g += stride) counts[g] = 0.f;
  if (i < 2) scalars[i] = 0.f;
}

// ---------------------------------------------------------------------------
// z_q output (== z_q_st numerically), commitment-loss partial sum,
// dw / counts segment sums via f32 atomics. Vectorized: 4 consecutive hw
// per thread -> b128 z-load, b128 idx-load, b128 z_q-store.
// ---------------------------------------------------------------------------
__global__ void vq_quant(const float* __restrict__ z, const float* __restrict__ w,
                         const int* __restrict__ idx, float* __restrict__ zq_out,
                         float* __restrict__ dw, float* __restrict__ counts,
                         float* __restrict__ loss_acc) {
  __shared__ float red[256];
  float acc = 0.f;
  const size_t nChunk = (size_t)TOTAL_Z / 4;
  for (size_t c = (size_t)blockIdx.x * blockDim.x + threadIdx.x;
       c < nChunk; c += (size_t)gridDim.x * blockDim.x) {
    size_t g = c * 4;
    int hw = (int)(g & 4095);
    int d  = (int)((g >> 12) & 255);
    int b  = (int)(g >> 20);
    int n  = b * 4096 + hw;
    int4   kk = *(const int4*)(idx + n);
    float4 zv = *(const float4*)(z + g);
    float4 wv;
    wv.x = w[(size_t)kk.x * 256 + d];
    wv.y = w[(size_t)kk.y * 256 + d];
    wv.z = w[(size_t)kk.z * 256 + d];
    wv.w = w[(size_t)kk.w * 256 + d];
    *(float4*)(zq_out + g) = wv;            // z_q_st == z_q numerically
    float dx = zv.x - wv.x, dy = zv.y - wv.y, dz = zv.z - wv.z, dwv = zv.w - wv.w;
    acc += dx * dx + dy * dy + dz * dz + dwv * dwv;
    atomicAdd(&dw[(size_t)kk.x * 256 + d], zv.x);
    atomicAdd(&dw[(size_t)kk.y * 256 + d], zv.y);
    atomicAdd(&dw[(size_t)kk.z * 256 + d], zv.z);
    atomicAdd(&dw[(size_t)kk.w * 256 + d], zv.w);
    if (d == 0) {
      atomicAdd(&counts[kk.x], 1.0f);
      atomicAdd(&counts[kk.y], 1.0f);
      atomicAdd(&counts[kk.z], 1.0f);
      atomicAdd(&counts[kk.w], 1.0f);
    }
  }
  red[threadIdx.x] = acc;
  __syncthreads();
  for (int s = 128; s > 0; s >>= 1) {
    if ((int)threadIdx.x < s) red[threadIdx.x] += red[threadIdx.x + s];
    __syncthreads();
  }
  if (threadIdx.x == 0) atomicAdd(loss_acc, red[0]);
}

// ---------------------------------------------------------------------------
// new_cluster_size + global sum n_total.
// ---------------------------------------------------------------------------
__global__ void vq_cluster(const float* __restrict__ ema_cs,
                           const float* __restrict__ counts,
                           float* __restrict__ ncs_out,
                           float* __restrict__ ntotal) {
  __shared__ float red[256];
  int k = blockIdx.x * 256 + threadIdx.x;
  float v = ema_cs[k] * 0.99f + 0.01f * counts[k];
  ncs_out[k] = v;
  red[threadIdx.x] = v;
  __syncthreads();
  for (int s = 128; s > 0; s >>= 1) {
    if ((int)threadIdx.x < s) red[threadIdx.x] += red[threadIdx.x + s];
    __syncthreads();
  }
  if (threadIdx.x == 0) atomicAdd(ntotal, red[0]);
}

// ---------------------------------------------------------------------------
// EMA-w, new_weight, indices-as-float, loss scalar.
// ---------------------------------------------------------------------------
__global__ void vq_finalize(const float* __restrict__ ema_w,
                            const float* __restrict__ dw,
                            const float* __restrict__ ncs,
                            const float* __restrict__ scalars,
                            const int* __restrict__ idx,
                            float* __restrict__ weight_out,
                            float* __restrict__ emaw_out,
                            float* __restrict__ idxf_out,
                            float* __restrict__ loss_out) {
  int g = blockIdx.x * 256 + threadIdx.x;
  int k = g >> 8;
  float ntot = scalars[0];
  float ncsk = ncs[k];
  float cs = (ncsk + 1e-5f) / (ntot + (float)K_CODES * 1e-5f) * ntot;
  float ne = ema_w[g] * 0.99f + 0.01f * dw[g];
  emaw_out[g] = ne;
  weight_out[g] = ne / cs;
  if (g < N_PIX) idxf_out[g] = (float)idx[g];
  if (g == 0) loss_out[0] = scalars[1] * (0.25f / (float)TOTAL_Z);
}

extern "C" void kernel_launch(void* const* d_in, const int* in_sizes, int n_in,
                              void* d_out, int out_size, void* d_ws, size_t ws_size,
                              hipStream_t stream) {
  const float* z       = (const float*)d_in[0];
  const float* weight  = (const float*)d_in[1];
  const float* ema_cs  = (const float*)d_in[2];
  const float* ema_w   = (const float*)d_in[3];
  float* out = (float*)d_out;
  char* ws = (char*)d_ws;

  unsigned short* zA = (unsigned short*)ws;                 // 33,554,432 B
  unsigned short* wB = (unsigned short*)(ws + 33554432);    //  2,097,152 B
  float* wnorm   = (float*)(ws + 35651584);                 //     16,384 B
  int*   idx     = (int*)  (ws + 35667968);                 //    262,144 B
  float* counts  = (float*)(ws + 35930112);                 //     16,384 B
  float* dw      = (float*)(ws + 35946496);                 //  4,194,304 B
  float* scalars = (float*)(ws + 40140800);                 //          8 B

  // Output tuple offsets (z_q, loss, indices, new_weight, new_cs, new_ema_w)
  float* o_zq   = out;
  float* o_loss = out + 16777216;
  float* o_idx  = out + 16777217;
  float* o_w    = out + 16842753;
  float* o_ncs  = out + 17891329;
  float* o_emaw = out + 17895425;

  vq_init       <<<1024, 256, 0, stream>>>(dw, counts, scalars);
  vq_swizzle_a  <<<M_TILES, 256, 0, stream>>>(z, zA);
  vq_swizzle_b  <<<K_TILES, 256, 0, stream>>>(weight, wB);
  vq_wnorm      <<<K_CODES / 8, 256, 0, stream>>>(weight, wnorm);
  vq_gemm_argmin<<<M_TILES / 16, 256, 0, stream>>>(zA, wB, wnorm, idx);
  vq_quant      <<<2048, 256, 0, stream>>>(z, weight, idx, o_zq, dw, counts, scalars + 1);
  vq_cluster    <<<K_CODES / 256, 256, 0, stream>>>(ema_cs, counts, o_ncs, scalars);
  vq_finalize   <<<(K_CODES * D_DIM) / 256, 256, 0, stream>>>(ema_w, dw, o_ncs, scalars,
                                                              idx, o_w, o_emaw, o_idx, o_loss);
}
